// Densetnt_83021717832217
// MI455X (gfx1250) — matrix operand order repaired and checked by reading, not compiled, not run on hardware
//
#include <hip/hip_runtime.h>
#include <hip/hip_bf16.h>
#include <math.h>

// ---------------------------------------------------------------------------
// DenseTNT head for MI455X (gfx1250, wave32).
// All GEMMs use v_wmma_f32_16x16x32_f16; activations chain through registers
// (D-layout -> next B-operand needs only one shfl_xor(16) per VGPR).
// Attention over NT=384 uses online (flash) softmax.
// ---------------------------------------------------------------------------

typedef _Float16 h16;
typedef __attribute__((ext_vector_type(16))) _Float16 v16h;
typedef __attribute__((ext_vector_type(8)))  float    v8f;

#define NG 16384
#define NT 384
#define CD 128
#define TK 150
#define NB 8

// workspace layout (h16 element offsets)
#define MAT_E   172032              // 10 matrices * 16384 + tdec1 8192
#define KV_E    49152               // 128*384
#define PLAIN_E (NB * 4 * KV_E)

// d_out float offsets: vals | traj_preds | pred_goals
#define OUT_TRAJ 1200
#define OUT_PG   73200

static constexpr float RSQRT_C = 0.08838834764831845f; // 1/sqrt(128)

struct Acc8 { v8f v[8]; };

static __device__ __forceinline__ v8f wmma16(v16h a, v16h b, v8f c) {
  return __builtin_amdgcn_wmma_f32_16x16x32_f16(false, a, false, b, (short)0, c,
                                                false, false);
}

static __device__ __forceinline__ void zero8(Acc8& a) {
#pragma unroll
  for (int m = 0; m < 8; ++m)
#pragma unroll
    for (int r = 0; r < 8; ++r) a.v[m][r] = 0.0f;
}

// acc += W(128x128, A-operand swizzled tiles) * B(4 ksteps in regs)
static __device__ __forceinline__ void gemm8(Acc8& acc, const h16* __restrict__ WA,
                                             const v16h B[4], int lane) {
#pragma unroll
  for (int s = 0; s < 4; ++s)
#pragma unroll
    for (int m = 0; m < 8; ++m) {
      v16h a = *(const v16h*)(WA + (((m * 4 + s) * 32 + lane) << 4));
      acc.v[m] = wmma16(a, B[s], acc.v[m]);
    }
}

// D layout: lane L holds channel M = 16*m + 8*(L/16) + r in acc.v[m][r]
static __device__ __forceinline__ void addbias(Acc8& a, const float* b, int hh) {
#pragma unroll
  for (int m = 0; m < 8; ++m)
#pragma unroll
    for (int r = 0; r < 8; ++r) a.v[m][r] += b[16 * m + 8 * hh + r];
}

static __device__ __forceinline__ void addbias_relu(Acc8& a, const float* b, int hh) {
#pragma unroll
  for (int m = 0; m < 8; ++m)
#pragma unroll
    for (int r = 0; r < 8; ++r)
      a.v[m][r] = fmaxf(a.v[m][r] + b[16 * m + 8 * hh + r], 0.0f);
}

// Repack channel-major accumulators into f16 B-operand for kstep s.
// B layout used here (contiguous k, as in the A tiles): lane L, half kk: K = 32*s + 16*(L/16) + kk.
// Missing 8 channels live in the partner lane (L xor 16) -> one shuffle/reg.
static __device__ __forceinline__ v16h xrepack(const Acc8& a, int s, int hh) {
  float other[8];
#pragma unroll
  for (int r = 0; r < 8; ++r) {
    float send = hh ? a.v[2 * s][r] : a.v[2 * s + 1][r];
    other[r] = __shfl_xor(send, 16, 32);
  }
  v16h out;
#pragma unroll
  for (int kk = 0; kk < 8; ++kk) {
    out[kk]     = (h16)(hh ? other[kk] : a.v[2 * s][kk]);
    out[8 + kk] = (h16)(hh ? a.v[2 * s + 1][kk] : other[kk]);
  }
  return out;
}

// goal(x,y) -> relu(W(128x2)*xy + b), produced directly as a B operand.
static __device__ __forceinline__ void stageA(v16h B0[4], float gx, float gy,
                                              const float* w0, const float* w1,
                                              const float* bb, int hh) {
#pragma unroll
  for (int s = 0; s < 4; ++s) {
    v16h t;
#pragma unroll
    for (int kk = 0; kk < 16; ++kk) {
      int ch = s * 32 + hh * 16 + kk;
      t[kk] = (h16)fmaxf(w0[ch] * gx + w1[ch] * gy + bb[ch], 0.0f);
    }
    B0[s] = t;
  }
}

// Flash-style attention: oacc = normalize(softmax(K^T q + mask)) * V, all WMMA.
static __device__ __forceinline__ void online_mha(Acc8& oacc, const v16h qB[4],
                                                  const h16* __restrict__ kA,
                                                  const h16* __restrict__ vA,
                                                  const float* lmask,
                                                  int lane, int hh) {
  zero8(oacc);
  float mrun = -3.0e38f, lrun = 0.0f;
  for (int bi = 0; bi < 12; ++bi) {
    __builtin_prefetch(kA + (((2 * (bi + 1)) * 4 * 32) << 4), 0, 1);
    v8f st[2];
    float bm = -3.0e38f;
#pragma unroll
    for (int tt = 0; tt < 2; ++tt) {
#pragma unroll
      for (int r = 0; r < 8; ++r) st[tt][r] = 0.0f;
#pragma unroll
      for (int s = 0; s < 4; ++s) {
        v16h a = *(const v16h*)(kA + ((((2 * bi + tt) * 4 + s) * 32 + lane) << 4));
        st[tt] = wmma16(a, qB[s], st[tt]);  // scoresT tile: M=t, N=goal
      }
#pragma unroll
      for (int r = 0; r < 8; ++r) {
        float v = st[tt][r] + lmask[(2 * bi + tt) * 16 + 8 * hh + r];
        st[tt][r] = v;
        bm = fmaxf(bm, v);
      }
    }
    bm = fmaxf(bm, __shfl_xor(bm, 16, 32));     // combine lane pair (same goal)
    float mnew = fmaxf(mrun, bm);
    float corr = __expf(mrun - mnew);
    float ps[2][8];
    float bs = 0.0f;
#pragma unroll
    for (int tt = 0; tt < 2; ++tt)
#pragma unroll
      for (int r = 0; r < 8; ++r) {
        float p = __expf(st[tt][r] - mnew);
        ps[tt][r] = p;
        bs += p;
      }
    bs += __shfl_xor(bs, 16, 32);
    lrun = lrun * corr + bs;
#pragma unroll
    for (int m = 0; m < 8; ++m)
#pragma unroll
      for (int r = 0; r < 8; ++r) oacc.v[m][r] *= corr;
    // pack probabilities into B operand (K = t within this 32-block)
    float other[8];
#pragma unroll
    for (int r = 0; r < 8; ++r) {
      float send = hh ? ps[0][r] : ps[1][r];
      other[r] = __shfl_xor(send, 16, 32);
    }
    v16h pb;
#pragma unroll
    for (int kk = 0; kk < 8; ++kk) {
      pb[kk]     = (h16)(hh ? other[kk] : ps[0][kk]);
      pb[8 + kk] = (h16)(hh ? ps[1][kk] : other[kk]);
    }
#pragma unroll
    for (int m = 0; m < 8; ++m) {
      v16h a = *(const v16h*)(vA + (((m * 12 + bi) * 32 + lane) << 4));
      oacc.v[m] = wmma16(a, pb, oacc.v[m]);
    }
    mrun = mnew;
  }
  float inv = 1.0f / lrun;
#pragma unroll
  for (int m = 0; m < 8; ++m)
#pragma unroll
    for (int r = 0; r < 8; ++r) oacc.v[m][r] *= inv;
}

// ---------------------------------------------------------------------------
// kW: swizzle weight matrices to f16 A-operand tile form.
// mats: 0 cat(:, :128) 1 gf_wq 2 gf_wk 3 gf_wv 4 gf_wo 5 tf_wq 6 tf_wk
//       7 tf_wv 8 tf_wo 9 tsg_w1 10 tdec(:, :128) (60 rows padded to 64)
// ---------------------------------------------------------------------------
__global__ __launch_bounds__(256) void kW_swz_weights(
    const float* __restrict__ catw, const float* __restrict__ w1,
    const float* __restrict__ w2, const float* __restrict__ w3,
    const float* __restrict__ w4, const float* __restrict__ w5,
    const float* __restrict__ w6, const float* __restrict__ w7,
    const float* __restrict__ w8, const float* __restrict__ w9,
    const float* __restrict__ tdecw, h16* __restrict__ matb) {
  int idx = blockIdx.x * 256 + threadIdx.x;
  if (idx >= MAT_E) return;
  int mat, e, off;
  if (idx < 163840) { mat = idx >> 14; e = idx & 16383; off = mat << 14; }
  else              { mat = 10; e = idx - 163840; off = 163840; }
  int tile = e >> 9, lane = (e >> 4) & 31, i = e & 15;
  int m = tile >> 2, s = tile & 3;
  int pr = i >> 1, half = i & 1, g = pr >> 2, j = pr & 3;
  int M = m * 16 + (lane & 15);                              // A: M = lane%16
  int K = s * 32 + (lane >> 4) * 16 + i;  // A K mapping
  float v = 0.0f;
  switch (mat) {
    case 0: v = catw[M * 256 + K]; break;
    case 1: v = w1[M * 128 + K]; break;
    case 2: v = w2[M * 128 + K]; break;
    case 3: v = w3[M * 128 + K]; break;
    case 4: v = w4[M * 128 + K]; break;
    case 5: v = w5[M * 128 + K]; break;
    case 6: v = w6[M * 128 + K]; break;
    case 7: v = w7[M * 128 + K]; break;
    case 8: v = w8[M * 128 + K]; break;
    case 9: v = w9[M * 128 + K]; break;
    default: v = (M < 60) ? tdecw[M * 256 + K] : 0.0f; break;
  }
  matb[off + e] = (h16)v;
}

// ---------------------------------------------------------------------------
// kP: per batch, K/V = W*traj + b (scale folded into K) via WMMA; cvec/dvec.
// ---------------------------------------------------------------------------
__global__ __launch_bounds__(256) void kP_kv(
    const float* __restrict__ traj, const float* __restrict__ graph,
    const float* __restrict__ catw, const float* __restrict__ catb,
    const float* __restrict__ tdecw, const float* __restrict__ tdecb,
    const float* __restrict__ gfbk, const float* __restrict__ gfbv,
    const float* __restrict__ tfbk, const float* __restrict__ tfbv,
    const h16* __restrict__ matb, h16* __restrict__ plainb,
    float* __restrict__ cvec, float* __restrict__ dvec) {
  int b = blockIdx.x;
  int tid = threadIdx.x, lane = tid & 31, wv = tid >> 5, hh = lane >> 4;
  const float* tb = traj + b * CD * NT;
  for (int tq = 0; tq < 3; ++tq) {
    int tt = wv * 3 + tq;                    // 8 waves * 3 = 24 t-tiles
    int t = tt * 16 + (lane & 15);
    v16h B[4];
#pragma unroll
    for (int s = 0; s < 4; ++s) {
      v16h x;
#pragma unroll
      for (int kk = 0; kk < 16; ++kk) {
        int c = s * 32 + hh * 16 + kk;       // B operand K mapping
        x[kk] = (h16)tb[c * NT + t];
      }
      B[s] = x;
    }
    for (int wi = 0; wi < 4; ++wi) {
      const h16* WA; const float* bias; float scale;
      if (wi == 0)      { WA = matb + 2 * 16384; bias = gfbk; scale = RSQRT_C; }
      else if (wi == 1) { WA = matb + 3 * 16384; bias = gfbv; scale = 1.0f; }
      else if (wi == 2) { WA = matb + 6 * 16384; bias = tfbk; scale = RSQRT_C; }
      else              { WA = matb + 7 * 16384; bias = tfbv; scale = 1.0f; }
      Acc8 a; zero8(a);
      gemm8(a, WA, B, lane);
      h16* dst = plainb + (b * 4 + wi) * KV_E;
#pragma unroll
      for (int m = 0; m < 8; ++m)
#pragma unroll
        for (int r = 0; r < 8; ++r) {
          int c = 16 * m + 8 * hh + r;
          dst[c * NT + t] = (h16)((a.v[m][r] + bias[c]) * scale);
        }
    }
  }
  if (tid < 128) {  // cvec = W_cat[:,128:]*graph + b_cat
    float s = catb[tid];
    for (int j = 0; j < 128; ++j) s += catw[tid * 256 + 128 + j] * graph[b * 128 + j];
    cvec[b * 128 + tid] = s;
  } else if (tid < 192) {  // dvec = tdec[:,128:]*graph + tdec_b (padded to 64)
    int o = tid - 128;
    float s = 0.0f;
    if (o < 60) {
      s = tdecb[o];
      for (int j = 0; j < 128; ++j) s += tdecw[o * 256 + 128 + j] * graph[b * 128 + j];
    }
    dvec[b * 64 + o] = s;
  }
}

// ---------------------------------------------------------------------------
// kS: re-swizzle K/V into A-operand tiles. arr 0/2: K^T tiles (M=t, K=chan),
// arr 1/3: V tiles (M=chan, K=t). grid(768, NB), 256 thr.
// ---------------------------------------------------------------------------
__global__ __launch_bounds__(256) void kS_swz_kv(const h16* __restrict__ plainb,
                                                 h16* __restrict__ swzb) {
  int b = blockIdx.y;
  int idx = blockIdx.x * 256 + threadIdx.x;  // < 4*KV_E
  int arr = idx / KV_E;
  int e = idx - arr * KV_E;
  int tile = e >> 9, lane = (e >> 4) & 31, i = e & 15;
  int pr = i >> 1, half = i & 1, g = pr >> 2, j = pr & 3;
  const h16* src = plainb + (b * 4 + arr) * KV_E;
  int c, t;
  if ((arr & 1) == 0) {
    int tt = tile >> 2, s = tile & 3;
    t = tt * 16 + (lane & 15);
    c = s * 32 + (lane >> 4) * 16 + i;
  } else {
    int m = tile / 12, bi = tile - m * 12;
    c = m * 16 + (lane & 15);
    t = bi * 32 + (lane >> 4) * 16 + i;
  }
  swzb[(b * 4 + arr) * KV_E + e] = src[c * NT + t];
}

// ---------------------------------------------------------------------------
// k1: fused per-goal pipeline -> 131072 goal scores. grid(128, NB), 256 thr.
// ---------------------------------------------------------------------------
__global__ __launch_bounds__(256) void k1_goal_scores(
    const float* __restrict__ g2d, const float* __restrict__ gmask,
    const float* __restrict__ inst,
    const float* __restrict__ gsgw, const float* __restrict__ gsgb,
    const float* __restrict__ gfbq, const float* __restrict__ gfbo,
    const float* __restrict__ gdecw, const float* __restrict__ gdecb,
    const h16* __restrict__ matb, const h16* __restrict__ swzb,
    const float* __restrict__ cvec, float* __restrict__ scores) {
  __shared__ float lw0[CD], lw1[CD], lb0[CD], lcv[CD], lbq[CD], lbo[CD],
                   lgd[CD], lmask[NT];
  int b = blockIdx.y, tid = threadIdx.x, lane = tid & 31, wv = tid >> 5,
      hh = lane >> 4;
  if (tid < CD) {
    lw0[tid] = gsgw[2 * tid]; lw1[tid] = gsgw[2 * tid + 1]; lb0[tid] = gsgb[tid];
    lcv[tid] = cvec[b * CD + tid]; lbq[tid] = gfbq[tid]; lbo[tid] = gfbo[tid];
    lgd[tid] = gdecw[tid];
  }
  for (int t = tid; t < NT; t += 256) lmask[t] = inst[b * NT + t];
  __syncthreads();

  int goal = blockIdx.x * 128 + wv * 16 + (lane & 15);
  float gx = g2d[b * 2 * NG + goal];
  float gy = g2d[b * 2 * NG + NG + goal];

  v16h B0[4];
  stageA(B0, gx, gy, lw0, lw1, lb0, hh);                 // gfeat0 as B operand
  Acc8 acc; zero8(acc);
  gemm8(acc, matb, B0, lane);                            // cat GEMM
  addbias_relu(acc, lcv, hh);                            // gfeat
  v16h Bg[4];
#pragma unroll
  for (int s = 0; s < 4; ++s) Bg[s] = xrepack(acc, s, hh);
  Acc8 qa; zero8(qa);
  gemm8(qa, matb + 1 * 16384, Bg, lane);                 // q = Wq*gfeat
  addbias(qa, lbq, hh);
  v16h qB[4];
#pragma unroll
  for (int s = 0; s < 4; ++s) qB[s] = xrepack(qa, s, hh);

  Acc8 oacc;
  online_mha(oacc, qB, swzb + (b * 4 + 0) * KV_E, swzb + (b * 4 + 1) * KV_E,
             lmask, lane, hh);

  v16h oB[4];
#pragma unroll
  for (int s = 0; s < 4; ++s) oB[s] = xrepack(oacc, s, hh);
  Acc8 fa; zero8(fa);
  gemm8(fa, matb + 4 * 16384, oB, lane);                 // fused = Wo*out
  addbias(fa, lbo, hh);
  float part = 0.0f;
#pragma unroll
  for (int m = 0; m < 8; ++m)
#pragma unroll
    for (int r = 0; r < 8; ++r) part += fa.v[m][r] * lgd[16 * m + 8 * hh + r];
  part += __shfl_xor(part, 16, 32);
  if (lane < 16)
    scores[b * NG + goal] = part + gdecb[0] + gmask[b * NG + goal];
}

// ---------------------------------------------------------------------------
// k2: exact top-150 (JAX tie-break: lower index first), writes vals+pred_goals
// ---------------------------------------------------------------------------
__global__ __launch_bounds__(256) void k2_topk(const float* __restrict__ g2d,
                                               float* __restrict__ scores,
                                               float* __restrict__ outp) {
  __shared__ float sv[256];
  __shared__ int si[256];
  int b = blockIdx.x, tid = threadIdx.x;
  for (int k = 0; k < TK; ++k) {
    float best = -3.4e38f; int bidx = NG;
    for (int i = tid; i < NG; i += 256) {
      float v = scores[b * NG + i];
      if (v > best) { best = v; bidx = i; }
    }
    sv[tid] = best; si[tid] = bidx;
    __syncthreads();
    for (int off = 128; off > 0; off >>= 1) {
      if (tid < off) {
        float v2 = sv[tid + off]; int i2 = si[tid + off];
        if (v2 > sv[tid] || (v2 == sv[tid] && i2 < si[tid])) {
          sv[tid] = v2; si[tid] = i2;
        }
      }
      __syncthreads();
    }
    if (tid == 0) {
      int idx = si[0];
      outp[b * TK + k] = sv[0];
      outp[OUT_PG + b * 300 + k]       = g2d[b * 2 * NG + idx];
      outp[OUT_PG + b * 300 + 150 + k] = g2d[b * 2 * NG + NG + idx];
      scores[b * NG + idx] = -3.4e38f;
      __threadfence_block();
    }
    __syncthreads();
  }
}

// ---------------------------------------------------------------------------
// k3: trajectory head on top-150 goals. grid(NB), 320 thr (10 waves of 16 q).
// ---------------------------------------------------------------------------
__global__ __launch_bounds__(320) void k3_traj(
    const float* __restrict__ inst,
    const float* __restrict__ tsgw0, const float* __restrict__ tsgb0,
    const float* __restrict__ tsgb1,
    const float* __restrict__ tfbq, const float* __restrict__ tfbo,
    const h16* __restrict__ matb, const h16* __restrict__ swzb,
    const float* __restrict__ dvec, float* __restrict__ outp) {
  __shared__ float lw0[CD], lw1[CD], lb0[CD], lb1[CD], lbq[CD], lbo[CD],
                   lmask[NT], ldv[64];
  int b = blockIdx.x, tid = threadIdx.x, lane = tid & 31, wv = tid >> 5,
      hh = lane >> 4;
  if (tid < CD) {
    lw0[tid] = tsgw0[2 * tid]; lw1[tid] = tsgw0[2 * tid + 1];
    lb0[tid] = tsgb0[tid]; lb1[tid] = tsgb1[tid];
    lbq[tid] = tfbq[tid]; lbo[tid] = tfbo[tid];
  }
  if (tid < 64) ldv[tid] = dvec[b * 64 + tid];
  for (int t = tid; t < NT; t += 320) lmask[t] = inst[b * NT + t];
  __syncthreads();

  int k = wv * 16 + (lane & 15);
  bool valid = k < TK;
  float gx = valid ? outp[OUT_PG + b * 300 + k] : 0.0f;
  float gy = valid ? outp[OUT_PG + b * 300 + 150 + k] : 0.0f;

  v16h B0[4];
  stageA(B0, gx, gy, lw0, lw1, lb0, hh);                 // relu(tsg_w0*goal+b0)
  Acc8 ta; zero8(ta);
  gemm8(ta, matb + 9 * 16384, B0, lane);                 // tsg_w1
  addbias_relu(ta, lb1, hh);                             // tfeat (kept: residual)
  v16h Bt[4];
#pragma unroll
  for (int s = 0; s < 4; ++s) Bt[s] = xrepack(ta, s, hh);
  Acc8 qa; zero8(qa);
  gemm8(qa, matb + 5 * 16384, Bt, lane);                 // tf_wq
  addbias(qa, lbq, hh);
  v16h qB[4];
#pragma unroll
  for (int s = 0; s < 4; ++s) qB[s] = xrepack(qa, s, hh);

  Acc8 oacc;
  online_mha(oacc, qB, swzb + (b * 4 + 2) * KV_E, swzb + (b * 4 + 3) * KV_E,
             lmask, lane, hh);

  v16h oB[4];
#pragma unroll
  for (int s = 0; s < 4; ++s) oB[s] = xrepack(oacc, s, hh);
  Acc8 fa; zero8(fa);
  gemm8(fa, matb + 8 * 16384, oB, lane);                 // tf_wo
  addbias(fa, lbo, hh);
#pragma unroll
  for (int m = 0; m < 8; ++m)
#pragma unroll
    for (int r = 0; r < 8; ++r) fa.v[m][r] += ta.v[m][r];  // + tfeat residual

  v16h fB[4];
#pragma unroll
  for (int s = 0; s < 4; ++s) fB[s] = xrepack(fa, s, hh);
  v8f d4[4];
#pragma unroll
  for (int mm = 0; mm < 4; ++mm)
#pragma unroll
    for (int r = 0; r < 8; ++r) d4[mm][r] = 0.0f;
#pragma unroll
  for (int s = 0; s < 4; ++s)
#pragma unroll
    for (int mm = 0; mm < 4; ++mm) {
      v16h a = *(const v16h*)(matb + 163840 + (((mm * 4 + s) * 32 + lane) << 4));
      d4[mm] = wmma16(a, fB[s], d4[mm]);                 // tdec[:, :128]
    }
#pragma unroll
  for (int mm = 0; mm < 4; ++mm)
#pragma unroll
    for (int r = 0; r < 8; ++r) {
      int o = 16 * mm + 8 * hh + r;
      if (o < 60 && valid)
        outp[OUT_TRAJ + b * 9000 + o * 150 + k] = d4[mm][r] + ldv[o];
    }
}

// ---------------------------------------------------------------------------
extern "C" void kernel_launch(void* const* d_in, const int* in_sizes, int n_in,
                              void* d_out, int out_size, void* d_ws,
                              size_t ws_size, hipStream_t stream) {
  (void)in_sizes; (void)n_in; (void)out_size; (void)ws_size;
  const float* g2d   = (const float*)d_in[0];
  const float* gmask = (const float*)d_in[1];
  const float* graph = (const float*)d_in[2];
  const float* traj  = (const float*)d_in[4];
  const float* inst  = (const float*)d_in[6];
  const float* gsgw  = (const float*)d_in[7];
  const float* gsgb  = (const float*)d_in[8];
  const float* catw  = (const float*)d_in[9];
  const float* catb  = (const float*)d_in[10];
  const float* gfwq  = (const float*)d_in[11];
  const float* gfwk  = (const float*)d_in[12];
  const float* gfwv  = (const float*)d_in[13];
  const float* gfwo  = (const float*)d_in[14];
  const float* gfbq  = (const float*)d_in[15];
  const float* gfbk  = (const float*)d_in[16];
  const float* gfbv  = (const float*)d_in[17];
  const float* gfbo  = (const float*)d_in[18];
  const float* tfwq  = (const float*)d_in[19];
  const float* tfwk  = (const float*)d_in[20];
  const float* tfwv  = (const float*)d_in[21];
  const float* tfwo  = (const float*)d_in[22];
  const float* tfbq  = (const float*)d_in[23];
  const float* tfbo  = (const float*)d_in[26];
  const float* tfbk  = (const float*)d_in[24];
  const float* tfbv  = (const float*)d_in[25];
  const float* gdecw = (const float*)d_in[27];
  const float* gdecb = (const float*)d_in[28];
  const float* tsgw0 = (const float*)d_in[29];
  const float* tsgb0 = (const float*)d_in[30];
  const float* tsgw1 = (const float*)d_in[31];
  const float* tsgb1 = (const float*)d_in[32];
  const float* tdecw = (const float*)d_in[33];
  const float* tdecb = (const float*)d_in[34];
  float* outp = (float*)d_out;

  h16* matb   = (h16*)d_ws;
  h16* plainb = matb + MAT_E;
  h16* swzb   = plainb + PLAIN_E;
  float* fbase = (float*)((char*)d_ws + (size_t)(MAT_E + 2 * PLAIN_E) * 2);
  float* cvec = fbase;
  float* dvec = fbase + 1024;
  float* scores = fbase + 1536;

  kW_swz_weights<<<672, 256, 0, stream>>>(catw, gfwq, gfwk, gfwv, gfwo, tfwq,
                                          tfwk, tfwv, tfwo, tsgw1, tdecw, matb);
  kP_kv<<<NB, 256, 0, stream>>>(traj, graph, catw, catb, tdecw, tdecb, gfbk,
                                gfbv, tfbk, tfbv, matb, plainb, cvec, dvec);
  kS_swz_kv<<<dim3(768, NB), 256, 0, stream>>>(plainb, swzb);
  k1_goal_scores<<<dim3(128, NB), 256, 0, stream>>>(
      g2d, gmask, inst, gsgw, gsgb, gfbq, gfbo, gdecw, gdecb, matb, swzb, cvec,
      scores);
  k2_topk<<<NB, 256, 0, stream>>>(g2d, scores, outp);
  k3_traj<<<NB, 320, 0, stream>>>(inst, tsgw0, tsgb0, tsgb1, tfbq, tfbo, matb,
                                  swzb, dvec, outp);
}
